// GraphEvolution_56444460204094
// MI455X (gfx1250) — matrix operand order repaired and checked by reading, not compiled, not run on hardware
//
#include <hip/hip_runtime.h>
#include <math.h>

// ---------------- problem constants ----------------
#define Mn      8192          // T*N
#define F0n     4
#define Pn      4
#define INCH    8
#define HIDn    32
#define HEADSn  8
#define DGn     256
#define OUTCH   4
#define En      131072
#define ETOT    (En + Mn)     // edges + self loops = 139264
#define NLn     6
#define MSGSn   3
#define EPSn    1e-5f
#define SLOPEn  0.2f

typedef _Float16 half_t;
typedef _Float16 v8h  __attribute__((ext_vector_type(8)));
typedef _Float16 v16h __attribute__((ext_vector_type(16)));
typedef float    v8f  __attribute__((ext_vector_type(8)));

__device__ __forceinline__ float gelu_erf(float x) {
    return 0.5f * x * (1.0f + erff(x * 0.70710678118654752f));
}

// float atomic max via int max / uint min trick (valid incl. -inf init)
__device__ __forceinline__ void atomicMaxF(float* addr, float val) {
    if (val >= 0.0f) atomicMax((int*)addr, __float_as_int(val));
    else             atomicMin((unsigned int*)addr, __float_as_uint(val));
}

// low 32 bits of a generic pointer to a __shared__ object = LDS byte offset
__device__ __forceinline__ unsigned lds_off(const void* p) {
    return (unsigned)(unsigned long long)p;
}

// async 16-byte memory -> LDS copy (CDNA5 GLOBAL_LOAD_ASYNC_TO_LDS_B128, ASYNCcnt)
__device__ __forceinline__ void async_copy16(unsigned lds_dst, const void* gsrc) {
    asm volatile("global_load_async_to_lds_b128 %0, %1, off"
                 :: "v"(lds_dst), "v"(gsrc) : "memory");
}
__device__ __forceinline__ void async_wait0() {
    asm volatile("s_wait_asynccnt 0" ::: "memory");
}

// -------- WMMA GEMM: C = act(A[M,K](f16) @ W[N,K](f16)^T + bias) --------
// A/W tiles DMA'd to LDS via async-to-LDS; f32 WMMA accumulation.
// Outputs: Cf (f32, may be null) and/or Ch (f16, may be null).
#define BM 64
#define BN 32
#define LDK 264   // padded K stride in halfs; 528B rows keep 16B alignment

__global__ __launch_bounds__(256)
void gemm_wmma(const half_t* __restrict__ A, const half_t* __restrict__ W,
               const float* __restrict__ bias, float* __restrict__ Cf,
               half_t* __restrict__ Ch, int M, int N, int K, int act)
{
    __shared__ half_t As[BM][LDK];
    __shared__ half_t Ws[BN][LDK];
    const int tid = threadIdx.x;
    const int bM  = blockIdx.x * BM;
    const int bN  = blockIdx.y * BN;

    // async-stage A tile (BM x K halfs) straight into LDS, 16B per lane-op
    for (int i = tid; i < BM * K / 8; i += 256) {
        int row = (i * 8) / K, col = (i * 8) % K;
        async_copy16(lds_off(&As[row][col]), A + (size_t)(bM + row) * K + col);
    }
    // async-stage W tile (BN x K halfs); row n of W == column n of B
    for (int i = tid; i < BN * K / 8; i += 256) {
        int row = (i * 8) / K, col = (i * 8) % K;
        async_copy16(lds_off(&Ws[row][col]), W + (size_t)(bN + row) * K + col);
    }
    async_wait0();
    __syncthreads();

    const int wave = tid >> 5, lane = tid & 31;
    const int half = lane & 15, hi = lane >> 4;
    const int wM = (wave & 3) * 16;      // 4 waves along M
    const int wN = (wave >> 2) * 16;     // 2 waves along N

    v8f acc = {};
    for (int k0 = 0; k0 < K; k0 += 32) {
        // A fragment: K split at 8-granularity per lane-half (ISA 16-bit A 16x32)
        v8h alo = *(const v8h*)&As[wM + half][k0 + hi * 8];
        v8h ahi = *(const v8h*)&As[wM + half][k0 + 16 + hi * 8];
        // B fragment: K split at 8-granularity per lane-half (same as A)
        v8h blo = *(const v8h*)&Ws[wN + half][k0 + hi * 8];
        v8h bhi = *(const v8h*)&Ws[wN + half][k0 + 16 + hi * 8];
        v16h a, b;
        #pragma unroll
        for (int j = 0; j < 8; ++j) {
            a[j] = alo[j]; a[j + 8] = ahi[j];
            b[j] = blo[j]; b[j + 8] = bhi[j];
        }
        acc = __builtin_amdgcn_wmma_f32_16x16x32_f16(false, a, false, b,
                                                     (short)0, acc, false, false);
    }
    #pragma unroll
    for (int r = 0; r < 8; ++r) {
        int row = bM + wM + hi * 8 + r;   // C layout: lanes16-31 hold M=8..15
        int col = bN + wN + half;
        float v = acc[r] + bias[col];
        if (act == 1)      v = v > 0.0f ? v : 0.0f;       // relu
        else if (act == 2) v = gelu_erf(v);               // gelu
        size_t idx = (size_t)row * N + col;
        if (Cf) Cf[idx] = v;
        if (Ch) Ch[idx] = (half_t)v;
    }
}

// ---------------- f32 -> f16 weight conversion ----------------
__global__ void wconv(const float* __restrict__ s, half_t* __restrict__ d, int n) {
    int i = blockIdx.x * blockDim.x + threadIdx.x;
    if (i < n) d[i] = (half_t)s[i];
}

// ---------------- encoder input ----------------
__global__ void encode(const float* __restrict__ x, const float* __restrict__ params,
                       const float* __restrict__ w, const float* __restrict__ bvec,
                       float* __restrict__ x_ext, float* __restrict__ h32,
                       half_t* __restrict__ h16)
{
    int i = blockIdx.x * blockDim.x + threadIdx.x;
    if (i >= Mn * HIDn) return;
    int row = i >> 5, o = i & 31;
    float s = bvec[o];
    #pragma unroll
    for (int k = 0; k < F0n; ++k) s += x[row * F0n + k] * w[o * INCH + k];
    #pragma unroll
    for (int k = 0; k < Pn; ++k)  s += params[k] * w[o * INCH + F0n + k];
    x_ext[i] = s;
    float g = gelu_erf(s);
    h32[i] = g;
    h16[i] = (half_t)g;
}

// -------- LayerNorm(h + t) in place (f32 master + f16 copy), one wave/row --------
__global__ __launch_bounds__(256)
void ln_residual(float* __restrict__ h32, half_t* __restrict__ h16,
                 const float* __restrict__ t, const float* __restrict__ g,
                 const float* __restrict__ b, int D)
{
    int wave = threadIdx.x >> 5, lane = threadIdx.x & 31;
    int row = blockIdx.x * 8 + wave;
    int n = D >> 5;
    float vals[8];
    float s = 0.0f;
    for (int i = 0; i < n; ++i) {
        int c = lane + (i << 5);
        vals[i] = h32[(size_t)row * D + c] + t[(size_t)row * D + c];
        s += vals[i];
    }
    for (int o = 16; o > 0; o >>= 1) s += __shfl_xor(s, o, 32);
    float mean = s / D;
    float v = 0.0f;
    for (int i = 0; i < n; ++i) { float d = vals[i] - mean; v += d * d; }
    for (int o = 16; o > 0; o >>= 1) v += __shfl_xor(v, o, 32);
    float inv = rsqrtf(v / D + EPSn);
    for (int i = 0; i < n; ++i) {
        int c = lane + (i << 5);
        float r = (vals[i] - mean) * inv * g[c] + b[c];
        h32[(size_t)row * D + c] = r;
        h16[(size_t)row * D + c] = (half_t)r;
    }
}

// ---------------- misc elementwise ----------------
__global__ void add_dual(float* __restrict__ a32, half_t* __restrict__ a16,
                         const float* __restrict__ b, int n) {
    int i = blockIdx.x * blockDim.x + threadIdx.x;
    if (i < n) { float v = a32[i] + b[i]; a32[i] = v; a16[i] = (half_t)v; }
}
__global__ void fill_f(float* __restrict__ p, float v, int n) {
    int i = blockIdx.x * blockDim.x + threadIdx.x;
    if (i < n) p[i] = v;
}
__global__ void reduce_sum(const float* __restrict__ a, int n, float* __restrict__ out) {
    int i = blockIdx.x * blockDim.x + threadIdx.x;
    float v = 0.0f;
    for (int j = i; j < n; j += gridDim.x * blockDim.x) v += a[j];
    for (int o = 16; o > 0; o >>= 1) v += __shfl_down(v, o, 32);
    if ((threadIdx.x & 31) == 0) atomicAdd(out, v);
}
__global__ void bias_gelu(const float* __restrict__ acc, const float* __restrict__ bias,
                          float* __restrict__ y32, half_t* __restrict__ y16, int n) {
    int i = blockIdx.x * blockDim.x + threadIdx.x;
    if (i < n) {
        float v = gelu_erf(acc[i] + bias[i % DGn]);
        y32[i] = v; y16[i] = (half_t)v;
    }
}

// ---------------- GATv2 edge kernels (f32 path) ----------------
__global__ __launch_bounds__(256)
void gat_logits(const int* __restrict__ ei, const float* __restrict__ eattr,
                const float* __restrict__ easum,
                const float* __restrict__ xl, const float* __restrict__ xr,
                const float* __restrict__ We, const float* __restrict__ att,
                float* __restrict__ logits, float* __restrict__ mx)
{
    int e = (blockIdx.x * 256 + threadIdx.x) >> 5;
    if (e >= ETOT) return;
    int lane = threadIdx.x & 31;
    int src, dst; float ea;
    if (e < En) { src = ei[e]; dst = ei[En + e]; ea = eattr[e]; }
    else        { src = dst = e - En; ea = easum[0] * (1.0f / En); }
    #pragma unroll
    for (int h = 0; h < HEADSn; ++h) {
        int c = h * 32 + lane;
        float g = xl[(size_t)src * DGn + c] + xr[(size_t)dst * DGn + c] + ea * We[c];
        float v = g > 0.0f ? g : SLOPEn * g;
        float t = v * att[c];
        for (int o = 16; o > 0; o >>= 1) t += __shfl_down(t, o, 32);
        if (lane == 0) {
            logits[(size_t)e * HEADSn + h] = t;
            atomicMaxF(&mx[dst * HEADSn + h], t);
        }
    }
}

__global__ void gat_expsum(const int* __restrict__ ei, const float* __restrict__ logits,
                           const float* __restrict__ mx, float* __restrict__ exb,
                           float* __restrict__ sm)
{
    int i = blockIdx.x * blockDim.x + threadIdx.x;
    if (i >= ETOT * HEADSn) return;
    int e = i >> 3, h = i & 7;
    int dst = (e < En) ? ei[En + e] : (e - En);
    float ex = expf(logits[i] - mx[dst * HEADSn + h]);
    exb[i] = ex;
    atomicAdd(&sm[dst * HEADSn + h], ex);
}

__global__ __launch_bounds__(256)
void gat_scatter(const int* __restrict__ ei, const float* __restrict__ exb,
                 const float* __restrict__ sm, const float* __restrict__ xl,
                 float* __restrict__ acc)
{
    int e = (blockIdx.x * 256 + threadIdx.x) >> 5;
    if (e >= ETOT) return;
    int lane = threadIdx.x & 31;
    int src, dst;
    if (e < En) { src = ei[e]; dst = ei[En + e]; } else { src = dst = e - En; }
    #pragma unroll
    for (int h = 0; h < HEADSn; ++h) {
        float alpha = exb[(size_t)e * HEADSn + h] / sm[dst * HEADSn + h];
        atomicAdd(&acc[(size_t)dst * DGn + h * 32 + lane],
                  alpha * xl[(size_t)src * DGn + h * 32 + lane]);
    }
}

// ---------------- final projection + means residual ----------------
__global__ void final_out(const float* __restrict__ y, const float* __restrict__ wd,
                          const float* __restrict__ bd, const float* __restrict__ x,
                          float* __restrict__ out)
{
    int i = blockIdx.x * blockDim.x + threadIdx.x;
    if (i >= Mn * OUTCH) return;
    int row = i >> 2, o = i & 3;
    float s = bd[o];
    for (int k = 0; k < DGn; ++k) s += y[(size_t)row * DGn + k] * wd[o * DGn + k];
    if (o < 2) s += x[row * F0n + o];   // xc[:, :2] == x[:, :2]
    out[i] = s;
}

// ---------------- host orchestration ----------------
extern "C" void kernel_launch(void* const* d_in, const int* in_sizes, int n_in,
                              void* d_out, int out_size, void* d_ws, size_t ws_size,
                              hipStream_t stream)
{
    (void)in_sizes; (void)n_in; (void)out_size; (void)ws_size;
    const float* x      = (const float*)d_in[0];
    const int*   ei     = (const int*)d_in[1];
    const float* eattr  = (const float*)d_in[2];
    const float* params = (const float*)d_in[3];
    const float* w_enc  = (const float*)d_in[4];
    const float* b_enc  = (const float*)d_in[5];
    const float* w_gatr = (const float*)d_in[78];
    const float* b_gatr = (const float*)d_in[79];
    const float* w_dec  = (const float*)d_in[209];
    const float* b_dec  = (const float*)d_in[210];

    // ---- workspace layout: f32 region then f16 region ----
    float* wsf   = (float*)d_ws;
    float* x_ext = wsf;  wsf += Mn * HIDn;
    float* h32   = wsf;  wsf += Mn * HIDn;
    float* t32b  = wsf;  wsf += Mn * HIDn;
    float* y32   = wsf;  wsf += (size_t)Mn * DGn;
    float* tAf   = wsf;  wsf += (size_t)Mn * DGn;   // xl / ln inputs
    float* tBf   = wsf;  wsf += (size_t)Mn * DGn;   // xr / ln inputs
    float* acc   = wsf;  wsf += (size_t)Mn * DGn;
    float* logit = wsf;  wsf += (size_t)ETOT * HEADSn;
    float* exb   = wsf;  wsf += (size_t)ETOT * HEADSn;
    float* mx    = wsf;  wsf += Mn * HEADSn;
    float* sm    = wsf;  wsf += Mn * HEADSn;
    float* easum = wsf;  wsf += 8;                  // padded for alignment

    half_t* wsh  = (half_t*)wsf;
    half_t* h16    = wsh;  wsh += Mn * HIDn;
    half_t* t32a16 = wsh;  wsh += Mn * HIDn;
    half_t* y16    = wsh;  wsh += (size_t)Mn * DGn;
    half_t* mem16  = wsh;  wsh += (size_t)Mn * DGn;
    half_t* tAh    = wsh;  wsh += (size_t)Mn * DGn; // decoder intermediate
    half_t* wcache = wsh;                            // f16 weight cache

    dim3 blk(256);
    size_t woff = 0;
    auto W16 = [&](const float* w, int n) -> half_t* {
        half_t* p = wcache + woff; woff += (size_t)n;
        wconv<<<(n + 255) / 256, blk, 0, stream>>>(w, p, n);
        return p;
    };
    auto GEMM = [&](const half_t* A, const half_t* W, const float* bias,
                    float* Cf, half_t* Ch, int M, int N, int K, int act) {
        dim3 grid(M / BM, N / BN);
        gemm_wmma<<<grid, blk, 0, stream>>>(A, W, bias, Cf, Ch, M, N, K, act);
    };

    // encoder front
    encode<<<(Mn * HIDn + 255) / 256, blk, 0, stream>>>(x, params, w_enc, b_enc,
                                                        x_ext, h32, h16);

    // 6 post-norm encoder layers (seq_len==1 attention == out_proj(v_proj(.)))
    for (int L = 0; L < NLn; ++L) {
        int bse = 6 + 12 * L;
        half_t* Wv = W16((const float*)d_in[bse + 0], HIDn * HIDn);
        const float* bv = (const float*)d_in[bse + 1];
        half_t* Wo = W16((const float*)d_in[bse + 2], HIDn * HIDn);
        const float* bo = (const float*)d_in[bse + 3];
        half_t* W1 = W16((const float*)d_in[bse + 4], HIDn * HIDn);
        const float* b1 = (const float*)d_in[bse + 5];
        half_t* W2 = W16((const float*)d_in[bse + 6], HIDn * HIDn);
        const float* b2 = (const float*)d_in[bse + 7];
        const float *g1 = (const float*)d_in[bse + 8],  *be1 = (const float*)d_in[bse + 9];
        const float *g2 = (const float*)d_in[bse + 10], *be2 = (const float*)d_in[bse + 11];
        GEMM(h16, Wv, bv, nullptr, t32a16, Mn, HIDn, HIDn, 0);
        GEMM(t32a16, Wo, bo, t32b, nullptr, Mn, HIDn, HIDn, 0);
        ln_residual<<<Mn / 8, blk, 0, stream>>>(h32, h16, t32b, g1, be1, HIDn);
        GEMM(h16, W1, b1, nullptr, t32a16, Mn, HIDn, HIDn, 1);
        GEMM(t32a16, W2, b2, t32b, nullptr, Mn, HIDn, HIDn, 0);
        ln_residual<<<Mn / 8, blk, 0, stream>>>(h32, h16, t32b, g2, be2, HIDn);
    }
    add_dual<<<(Mn * HIDn + 255) / 256, blk, 0, stream>>>(h32, h16, x_ext, Mn * HIDn);
    GEMM(h16, W16(w_gatr, DGn * HIDn), b_gatr, y32, y16, Mn, DGn, HIDn, 0);
    hipMemcpyAsync(mem16, y16, sizeof(half_t) * (size_t)Mn * DGn,
                   hipMemcpyDeviceToDevice, stream);

    // edge_attr mean (for self-loop fill)
    hipMemsetAsync(easum, 0, sizeof(float), stream);
    reduce_sum<<<256, blk, 0, stream>>>(eattr, En, easum);

    // 3 GATv2 message-passing layers
    for (int g = 0; g < MSGSn; ++g) {
        int bse = 80 + 7 * g;
        half_t* Wl = W16((const float*)d_in[bse + 0], DGn * DGn);
        const float* bl = (const float*)d_in[bse + 1];
        half_t* Wr = W16((const float*)d_in[bse + 2], DGn * DGn);
        const float* br = (const float*)d_in[bse + 3];
        const float *We  = (const float*)d_in[bse + 4];
        const float *att = (const float*)d_in[bse + 5];
        const float *bia = (const float*)d_in[bse + 6];
        GEMM(y16, Wl, bl, tAf, nullptr, Mn, DGn, DGn, 0);   // xl (f32 for edge math)
        GEMM(y16, Wr, br, tBf, nullptr, Mn, DGn, DGn, 0);   // xr
        fill_f<<<(Mn * HEADSn + 255) / 256, blk, 0, stream>>>(mx, -INFINITY, Mn * HEADSn);
        hipMemsetAsync(sm, 0, sizeof(float) * Mn * HEADSn, stream);
        hipMemsetAsync(acc, 0, sizeof(float) * (size_t)Mn * DGn, stream);
        gat_logits<<<ETOT / 8, blk, 0, stream>>>(ei, eattr, easum, tAf, tBf, We, att, logit, mx);
        gat_expsum<<<(ETOT * HEADSn + 255) / 256, blk, 0, stream>>>(ei, logit, mx, exb, sm);
        gat_scatter<<<ETOT / 8, blk, 0, stream>>>(ei, exb, sm, tAf, acc);
        bias_gelu<<<(Mn * DGn + 255) / 256, blk, 0, stream>>>(acc, bia, y32, y16, Mn * DGn);
    }

    // 6 post-norm decoder layers (self-attn, cross-attn on mem, ffn)
    for (int L = 0; L < NLn; ++L) {
        int bse = 101 + 18 * L;
        half_t* sWv = W16((const float*)d_in[bse + 0], DGn * DGn);
        const float* sbv = (const float*)d_in[bse + 1];
        half_t* sWo = W16((const float*)d_in[bse + 2], DGn * DGn);
        const float* sbo = (const float*)d_in[bse + 3];
        half_t* cWv = W16((const float*)d_in[bse + 4], DGn * DGn);
        const float* cbv = (const float*)d_in[bse + 5];
        half_t* cWo = W16((const float*)d_in[bse + 6], DGn * DGn);
        const float* cbo = (const float*)d_in[bse + 7];
        half_t* W1 = W16((const float*)d_in[bse + 8], DGn * DGn);
        const float* b1 = (const float*)d_in[bse + 9];
        half_t* W2 = W16((const float*)d_in[bse + 10], DGn * DGn);
        const float* b2 = (const float*)d_in[bse + 11];
        const float *g1  = (const float*)d_in[bse + 12], *be1 = (const float*)d_in[bse + 13];
        const float *g2  = (const float*)d_in[bse + 14], *be2 = (const float*)d_in[bse + 15];
        const float *g3  = (const float*)d_in[bse + 16], *be3 = (const float*)d_in[bse + 17];
        GEMM(y16, sWv, sbv, nullptr, tAh, Mn, DGn, DGn, 0);
        GEMM(tAh, sWo, sbo, tBf, nullptr, Mn, DGn, DGn, 0);
        ln_residual<<<Mn / 8, blk, 0, stream>>>(y32, y16, tBf, g1, be1, DGn);
        GEMM(mem16, cWv, cbv, nullptr, tAh, Mn, DGn, DGn, 0);
        GEMM(tAh, cWo, cbo, tBf, nullptr, Mn, DGn, DGn, 0);
        ln_residual<<<Mn / 8, blk, 0, stream>>>(y32, y16, tBf, g2, be2, DGn);
        GEMM(y16, W1, b1, nullptr, tAh, Mn, DGn, DGn, 1);
        GEMM(tAh, W2, b2, tBf, nullptr, Mn, DGn, DGn, 0);
        ln_residual<<<Mn / 8, blk, 0, stream>>>(y32, y16, tBf, g3, be3, DGn);
    }

    final_out<<<(Mn * OUTCH + 255) / 256, blk, 0, stream>>>(y32, w_dec, b_dec, x,
                                                            (float*)d_out);
}
